// EdgeDecoder_44186623541776
// MI455X (gfx1250) — matrix operand order repaired and checked by reading, not compiled, not run on hardware
//
#include <hip/hip_runtime.h>
#include <hip/hip_bf16.h>

#define H 128

typedef __attribute__((ext_vector_type(16))) __bf16 v16bf;
typedef __attribute__((ext_vector_type(8)))  float  v8f;
typedef __attribute__((ext_vector_type(4)))  float  v4f;

// Kernel 1: W1 (256x128 f32, row-major [k][n]) -> W1T bf16 [n][k] in workspace.
__global__ __launch_bounds__(256) void w1_to_bf16_t(const float* __restrict__ W1,
                                                    __bf16* __restrict__ w1t) {
  int i = blockIdx.x * blockDim.x + threadIdx.x;
  if (i >= 2 * H * H) return;
  int n = i / (2 * H);      // 0..127  (output column)
  int k = i % (2 * H);      // 0..255  (concat feature index)
  w1t[n * (2 * H) + k] = (__bf16)W1[k * H + n];
}

// Kernel 2: fused gather + [E,256]x[256,128] bf16 WMMA GEMM + relu + W2 dot.
// 256 threads = 8 waves; each wave owns 32 edges (two 16-row M-tiles) so that
// every B fragment fetched feeds 2 WMMAs (raises FLOPs per VMEM issue).
// Block covers 256 edges.
__global__ __launch_bounds__(256) void edge_decoder_wmma(
    const float* __restrict__ z_user, const float* __restrict__ z_item,
    const int* __restrict__ row, const int* __restrict__ col,
    const __bf16* __restrict__ w1t, const float* __restrict__ b1,
    const float* __restrict__ W2, const float* __restrict__ b2,
    float* __restrict__ out, int E)
{
  const int lane = threadIdx.x & 31;
  const int wave = threadIdx.x >> 5;
  const int nloc = lane & 15;   // column within 16-wide tile / M within M-tile
  const int hi   = lane >> 4;   // 0: lanes 0-15, 1: lanes 16-31

  // Warm L2/L0 path for the shared bf16 weights (emits global_prefetch_b8).
  __builtin_prefetch(w1t + (threadIdx.x << 4), 0, 0);

  const int eBase = blockIdx.x * 256 + wave * 32;

  // Per-lane edge indices for the two M-groups.
  const float* urow[2];
  const float* irow[2];
  #pragma unroll
  for (int g = 0; g < 2; ++g) {
    const int e  = eBase + g * 16 + nloc;
    const int eg = (e < E) ? e : (E - 1);        // clamp gathers; stores guarded
    urow[g] = z_user + (long)row[eg] * H;
    irow[g] = z_item + (long)col[eg] * H;
  }

  v8f c[2][8];
  #pragma unroll
  for (int g = 0; g < 2; ++g)
    #pragma unroll
    for (int t = 0; t < 8; ++t) c[g][t] = v8f{};

  // K loop: 8 steps of 32 over the concatenated 256-wide feature.
  #pragma unroll
  for (int ks = 0; ks < 8; ++ks) {
    const int k0 = ks * 32;
    const int kA = hi ? 16 : 0;

    // A fragments (16x32 bf16): lane L holds M=L%16.
    // lanes 0-15: K = k0+{0..15}; lanes 16-31: K = k0+{16..31}.
    v16bf a[2];
    #pragma unroll
    for (int g = 0; g < 2; ++g) {
      const float* src = (k0 < H) ? (urow[g] + k0) : (irow[g] + (k0 - H));
      v4f f0 = *(const v4f*)(src + kA + 0);
      v4f f1 = *(const v4f*)(src + kA + 4);
      v4f f2 = *(const v4f*)(src + kA + 8);
      v4f f3 = *(const v4f*)(src + kA + 12);
      a[g][0]  = (__bf16)f0[0]; a[g][1]  = (__bf16)f0[1];
      a[g][2]  = (__bf16)f0[2]; a[g][3]  = (__bf16)f0[3];
      a[g][4]  = (__bf16)f1[0]; a[g][5]  = (__bf16)f1[1];
      a[g][6]  = (__bf16)f1[2]; a[g][7]  = (__bf16)f1[3];
      a[g][8]  = (__bf16)f2[0]; a[g][9]  = (__bf16)f2[1];
      a[g][10] = (__bf16)f2[2]; a[g][11] = (__bf16)f2[3];
      a[g][12] = (__bf16)f3[0]; a[g][13] = (__bf16)f3[1];
      a[g][14] = (__bf16)f3[2]; a[g][15] = (__bf16)f3[3];
    }

    // N loop: 8 tiles of 16 columns; B fragment is 32 contiguous bf16 of W1T[n].
    // lanes 0-15: K = k0..k0+15; lanes 16-31: K = k0+16..k0+31 (pairs per VGPR).
    const int kB = k0 + (hi ? 16 : 0);
    #pragma unroll
    for (int t = 0; t < 8; ++t) {
      const int n = t * 16 + nloc;
      v16bf bfrag = *(const v16bf*)(w1t + n * (2 * H) + kB);
      #pragma unroll
      for (int g = 0; g < 2; ++g) {
        c[g][t] = __builtin_amdgcn_wmma_f32_16x16x32_bf16(
            /*neg_a=*/false, a[g], /*neg_b=*/false, bfrag,
            /*c_mod=*/(short)0, c[g][t], /*reuse_a=*/false, /*reuse_b=*/false);
      }
    }
  }

  // Epilogue: C layout is lane -> N = lane%16, VGPR r -> M = r + 8*hi.
  // h = relu(c + b1[n]); partial = sum_t h * W2[n]; reduce across the 16 lanes
  // of each half (xor masks 1,2,4,8 stay within a half in wave32).
  float b1v[8], w2v[8];
  #pragma unroll
  for (int t = 0; t < 8; ++t) {
    b1v[t] = b1[t * 16 + nloc];
    w2v[t] = W2[t * 16 + nloc];
  }
  const float bb = b2[0];

  #pragma unroll
  for (int g = 0; g < 2; ++g) {
    float acc[8];
    #pragma unroll
    for (int r = 0; r < 8; ++r) {
      float s = 0.f;
      #pragma unroll
      for (int t = 0; t < 8; ++t) {
        float hvl = c[g][t][r] + b1v[t];
        hvl = hvl > 0.f ? hvl : 0.f;
        s += hvl * w2v[t];
      }
      acc[r] = s;
    }
    #pragma unroll
    for (int m = 1; m <= 8; m <<= 1) {
      #pragma unroll
      for (int r = 0; r < 8; ++r) acc[r] += __shfl_xor(acc[r], m, 32);
    }
    if (nloc == 0) {
      const int base = eBase + g * 16 + hi * 8;   // M = r + 8*hi
      #pragma unroll
      for (int r = 0; r < 8; ++r) {
        const int ee = base + r;
        if (ee < E) out[ee] = acc[r] + bb;
      }
    }
  }
}

extern "C" void kernel_launch(void* const* d_in, const int* in_sizes, int n_in,
                              void* d_out, int out_size, void* d_ws, size_t ws_size,
                              hipStream_t stream) {
  const float* z_user = (const float*)d_in[0];
  const float* z_item = (const float*)d_in[1];
  const int*   row    = (const int*)d_in[2];
  const int*   col    = (const int*)d_in[3];
  const float* W1     = (const float*)d_in[4];
  const float* b1     = (const float*)d_in[5];
  const float* W2     = (const float*)d_in[6];
  const float* b2     = (const float*)d_in[7];
  float*       out    = (float*)d_out;
  const int    E      = in_sizes[2];

  __bf16* w1t = (__bf16*)d_ws;   // 2*H*H bf16 = 64 KB of scratch

  w1_to_bf16_t<<<dim3((2 * H * H + 255) / 256), dim3(256), 0, stream>>>(W1, w1t);

  const int blocks = (E + 255) / 256;
  edge_decoder_wmma<<<dim3(blocks), dim3(256), 0, stream>>>(
      z_user, z_item, row, col, w1t, b1, W2, b2, out, E);
}
